// LaCTSWIGLULayer_5970004542302
// MI455X (gfx1250) — compile-verified, matrix operand order checked by reading, not run on hardware
//
#include <hip/hip_runtime.h>
#include <hip/hip_bf16.h>

typedef __bf16 v16bf __attribute__((ext_vector_type(16)));
typedef float  v8f   __attribute__((ext_vector_type(8)));

#define H_DIM 2048
#define NHEADS 16
#define HD 128
#define NFW 4
#define FD 512
#define CHUNKN 512
#define SEQ 2048
#define BATCH 2
#define NTOK (BATCH*SEQ)   /* 4096 */
#define NBH  (BATCH*NFW)   /* 8    */

// ---------------------------------------------------------------------------
// helpers
// ---------------------------------------------------------------------------
__device__ __forceinline__ __bf16 to_bf16(float f) {
    unsigned u = __builtin_bit_cast(unsigned, f);
    unsigned r = u + 0x7FFFu + ((u >> 16) & 1u);   // round-to-nearest-even
    unsigned short h = (unsigned short)(r >> 16);
    return __builtin_bit_cast(__bf16, h);
}

// CDNA5 16-bit A/B operand layout (16x32 per wave):
// lanes 0-15: row = lane,    K = 0..7  and 16..23
// lanes 16-31: row = lane-16, K = 8..15 and 24..31
__device__ __forceinline__ v16bf load_frag(const __bf16* base, int ld, int lane) {
    int row = lane & 15;
    int hi  = (lane >> 4) & 1;
    const __bf16* p = base + row * ld + hi * 8;
    v16bf r;
#pragma unroll
    for (int i = 0; i < 8; ++i) r[i] = p[i];
#pragma unroll
    for (int i = 0; i < 8; ++i) r[i + 8] = p[16 + i];
    return r;
}

__device__ __forceinline__ v8f wmma_bf16(v16bf a, v16bf b, v8f c) {
    return __builtin_amdgcn_wmma_f32_16x16x32_bf16(false, a, false, b,
                                                   (short)0, c, false, false);
}

__device__ __forceinline__ float siluf(float x) { return x / (1.f + __expf(-x)); }

// ---------------------------------------------------------------------------
// Generic batched WMMA GEMM:  C[m,n] (+)= sum_k A[m,k] * B[n,k]
// A element at A + m*sA0 + k*sA1 ; operands are either row-major (s1==1,
// vectorized float4 loads along K) or column-major (s0==1, vectorized float4
// loads along M).  K must be a multiple of 32; M a multiple of 128.
// GUARD adds row/col bounds handling (needed only for N=12 lr GEMM).
// 128x128 tile, BK=32, 256 threads = 8 waves (4x2), each wave 32x64.
// ---------------------------------------------------------------------------
#define BM 128
#define BN 128
#define BKD 32
#define LDP 8

__device__ __forceinline__ void store_bf8(__bf16* dst, const float* s) {
    union { uint4 u; __bf16 h[8]; } pk;
#pragma unroll
    for (int i = 0; i < 8; ++i) pk.h[i] = to_bf16(s[i]);
    *(uint4*)dst = pk.u;
}

// row-major source: element = src[(t0+row)*s0 + kb + c]
template<bool GUARD>
__device__ __forceinline__ void load_tile_row(
    __bf16 (*dst)[BKD + LDP], const float* __restrict__ src, long long s0,
    int t0, int kb, int limit, int tid)
{
    const int row = tid >> 1;
    const int c0  = (tid & 1) * 16;
    float buf[16];
    if (!GUARD || (t0 + row) < limit) {
        const float* p = src + (long long)(t0 + row) * s0 + kb + c0;
        const float4* p4 = (const float4*)p;
        float4 v0 = p4[0], v1 = p4[1], v2 = p4[2], v3 = p4[3];
        buf[0]=v0.x; buf[1]=v0.y; buf[2]=v0.z; buf[3]=v0.w;
        buf[4]=v1.x; buf[5]=v1.y; buf[6]=v1.z; buf[7]=v1.w;
        buf[8]=v2.x; buf[9]=v2.y; buf[10]=v2.z; buf[11]=v2.w;
        buf[12]=v3.x; buf[13]=v3.y; buf[14]=v3.z; buf[15]=v3.w;
        __builtin_prefetch(p + BKD, 0, 1);       // next K tile -> global_prefetch
    } else {
#pragma unroll
        for (int i = 0; i < 16; ++i) buf[i] = 0.f;
    }
    store_bf8(&dst[row][c0], buf);
    store_bf8(&dst[row][c0 + 8], buf + 8);
}

// column-major source: element = src[(t0+m) + (kb+k)*s1]  (never guarded here)
__device__ __forceinline__ void load_tile_col(
    __bf16 (*dst)[BKD + LDP], const float* __restrict__ src, long long s1,
    int t0, int kb, int tid)
{
    const int m0 = (tid & 31) * 4;
    const int kk = tid >> 5;
#pragma unroll
    for (int pass = 0; pass < 4; ++pass) {
        int k = kk + pass * 8;
        const float* p = src + (long long)(kb + k) * s1 + t0 + m0;
        float4 v = *(const float4*)p;
        dst[m0 + 0][k] = to_bf16(v.x);
        dst[m0 + 1][k] = to_bf16(v.y);
        dst[m0 + 2][k] = to_bf16(v.z);
        dst[m0 + 3][k] = to_bf16(v.w);
        __builtin_prefetch(p + (long long)BKD * s1, 0, 1);
    }
}

template<bool GUARD>
__global__ __launch_bounds__(256) void wmma_gemm_t(
    const float* __restrict__ A, long long sA0, long long sA1, long long batchA,
    const float* __restrict__ Bm, long long sB0, long long sB1, long long batchB,
    float* __restrict__ C, long long ldc, long long batchC,
    int M, int N, int K, int accumulate)
{
    __shared__ __bf16 As[BM][BKD + LDP];
    __shared__ __bf16 Bs[BN][BKD + LDP];

    const int bz = blockIdx.z;
    A  += (long long)bz * batchA;
    Bm += (long long)bz * batchB;
    C  += (long long)bz * batchC;

    const int tm = blockIdx.y * BM;
    const int tn = blockIdx.x * BN;
    const int tid  = threadIdx.x;
    const int lane = tid & 31;
    const int wave = tid >> 5;
    const int wm = wave & 3;        // 4 x 32 rows
    const int wn = wave >> 2;       // 2 x 64 cols

    v8f acc[2][4];
#pragma unroll
    for (int i = 0; i < 2; ++i)
#pragma unroll
        for (int j = 0; j < 4; ++j)
#pragma unroll
            for (int e = 0; e < 8; ++e) acc[i][j][e] = 0.f;

    for (int kb = 0; kb < K; kb += BKD) {
        if (sA1 == 1) load_tile_row<GUARD>(As, A, sA0, tm, kb, M, tid);
        else          load_tile_col(As, A, sA1, tm, kb, tid);
        if (sB1 == 1) load_tile_row<GUARD>(Bs, Bm, sB0, tn, kb, N, tid);
        else          load_tile_col(Bs, Bm, sB1, tn, kb, tid);
        __syncthreads();

        v16bf af[2], bfg[4];
#pragma unroll
        for (int i = 0; i < 2; ++i) af[i] = load_frag(&As[wm * 32 + i * 16][0], BKD + LDP, lane);
#pragma unroll
        for (int j = 0; j < 4; ++j) bfg[j] = load_frag(&Bs[wn * 64 + j * 16][0], BKD + LDP, lane);
#pragma unroll
        for (int i = 0; i < 2; ++i)
#pragma unroll
            for (int j = 0; j < 4; ++j)
                acc[i][j] = wmma_bf16(af[i], bfg[j], acc[i][j]);
        __syncthreads();
    }

    const int rlo = lane & 15;
    const int hi  = lane >> 4;
#pragma unroll
    for (int i = 0; i < 2; ++i)
#pragma unroll
        for (int j = 0; j < 4; ++j) {
            int col = tn + wn * 64 + j * 16 + rlo;
            if (GUARD && col >= N) continue;
#pragma unroll
            for (int r = 0; r < 8; ++r) {
                int row = tm + wm * 32 + i * 16 + hi * 8 + r;
                if (GUARD && row >= M) continue;
                long long idx = (long long)row * ldc + col;
                if (accumulate) C[idx] += acc[i][j][r];
                else            C[idx]  = acc[i][j][r];
            }
        }
}

// ---------------------------------------------------------------------------
// Post-QKV: rmsnorm(q,k), silu gates, per-head l2norm, rope, head reshapes.
// One block per token.
// ---------------------------------------------------------------------------
__global__ __launch_bounds__(256) void post_qkv_kernel(
    const float* __restrict__ qkv,
    const float* __restrict__ qnw, const float* __restrict__ knw,
    const float* __restrict__ qks, const float* __restrict__ qko,
    float* __restrict__ fq, float* __restrict__ fk, float* __restrict__ fv,
    float* __restrict__ qa, float* __restrict__ ka, float* __restrict__ va)
{
    const int t = blockIdx.x;
    const int b = t / SEQ, s = t % SEQ;
    const int tid = threadIdx.x;
    const float* row = qkv + (long long)t * (3 * H_DIM);

    __shared__ float sq[H_DIM];
    __shared__ float sk[H_DIM];
    __shared__ float red[256];
    __shared__ float hsum[8];

    const int i0 = tid * 8;
    float lq = 0.f, lk = 0.f;
#pragma unroll
    for (int u = 0; u < 8; ++u) {
        float qv = row[i0 + u];
        float kv = row[H_DIM + i0 + u];
        sq[i0 + u] = qv; sk[i0 + u] = kv;
        lq += qv * qv; lk += kv * kv;
    }
    red[tid] = lq; __syncthreads();
    for (int st = 128; st > 0; st >>= 1) { if (tid < st) red[tid] += red[tid + st]; __syncthreads(); }
    float sumq = red[0]; __syncthreads();
    red[tid] = lk; __syncthreads();
    for (int st = 128; st > 0; st >>= 1) { if (tid < st) red[tid] += red[tid + st]; __syncthreads(); }
    float sumk = red[0];
    if (tid < 8) hsum[tid] = 0.f;
    __syncthreads();

    const float rq = rsqrtf(sumq * (1.f / H_DIM) + 1e-6f);
    const float rk = rsqrtf(sumk * (1.f / H_DIM) + 1e-6f);

    float fqr[8], fkr[8];
    float lsq = 0.f, lsk = 0.f;
#pragma unroll
    for (int u = 0; u < 8; ++u) {
        int i = i0 + u;
        float qn = sq[i] * rq * qnw[i];
        float kn = sk[i] * rk * knw[i];
        float vv = row[2 * H_DIM + i];
        sq[i] = qn; sk[i] = kn;                 // keep for rope
        float fqv = siluf(qn * qks[2 * i]     + qko[2 * i]);
        float fkv = siluf(kn * qks[2 * i + 1] + qko[2 * i + 1]);
        float fvv = siluf(vv);
        fqr[u] = fqv; fkr[u] = fkv;
        lsq += fqv * fqv; lsk += fkv * fkv;
        int h4 = i >> 9, d4 = i & 511;
        fv[(((long long)(b * NFW + h4)) * SEQ + s) * FD + d4] = fvv;
        int h16 = i >> 7, d16 = i & 127;
        va[(((long long)(b * NHEADS + h16)) * SEQ + s) * HD + d16] = vv;
    }
    const int myh = tid >> 6;
    atomicAdd(&hsum[myh], lsq);
    atomicAdd(&hsum[4 + myh], lsk);
    __syncthreads();

    const float sfq = rsqrtf(hsum[myh] + 1e-12f);
    const float sfk = rsqrtf(hsum[4 + myh] + 1e-12f);
#pragma unroll
    for (int u = 0; u < 8; ++u) {
        int i = i0 + u;
        int h4 = i >> 9, d4 = i & 511;
        long long o = (((long long)(b * NFW + h4)) * SEQ + s) * FD + d4;
        fq[o] = fqr[u] * sfq;
        fk[o] = fkr[u] * sfk;
        // rope for attention path
        int h16 = i >> 7, hd = i & 127, dd = hd & 63;
        float fr = (float)s * __expf(-(float)dd * 0.20503693f);  // ln(5e5)/64
        float sn, cs; __sincosf(fr, &sn, &cs);
        float oq, ok;
        if (hd < 64) { oq = sq[i] * cs - sq[i + 64] * sn; ok = sk[i] * cs - sk[i + 64] * sn; }
        else         { oq = sq[i] * cs + sq[i - 64] * sn; ok = sk[i] * cs + sk[i - 64] * sn; }
        long long oa = (((long long)(b * NHEADS + h16)) * SEQ + s) * HD + hd;
        qa[oa] = oq; ka[oa] = ok;
    }
}

// ---------------------------------------------------------------------------
// lr = softplus(x@lr_w.T + lr_b + BASE_LR_INV), layout [3][B][NFW][SEQ]
// ---------------------------------------------------------------------------
__global__ void lr_post_kernel(const float* __restrict__ lrpre,
                               const float* __restrict__ lrb,
                               float* __restrict__ lrout)
{
    int i = blockIdx.x * 256 + threadIdx.x;
    const int n = 3 * BATCH * NFW * SEQ;
    if (i >= n) return;
    int s = i % SEQ; int rest = i / SEQ;
    int h = rest % NFW; rest /= NFW;
    int b = rest % BATCH; int t = rest / BATCH;
    int col = t * NFW + h;
    float x = lrpre[((long long)(b * SEQ + s)) * 12 + col] + lrb[col] + (-6.9072554f);
    lrout[i] = (x > 20.f) ? x : log1pf(__expf(x));
}

// ---------------------------------------------------------------------------
// broadcast initial fast weights over batch
// ---------------------------------------------------------------------------
__global__ void bcast_w_kernel(const float* __restrict__ src, float* __restrict__ dst)
{
    const long long per = (long long)FD * FD;
    const long long n = (long long)NBH * per;
    for (long long i = (long long)blockIdx.x * 256 + threadIdx.x; i < n;
         i += (long long)gridDim.x * 256) {
        int bh = (int)(i / per); int h = bh % NFW;
        dst[i] = src[(long long)h * per + (i % per)];
    }
}

// ---------------------------------------------------------------------------
// SWIGLU mid: hq->a1, hidden->g1, l0*dh1->g2, l2*dh2->dh, l1*v->a2 (in-place)
// ---------------------------------------------------------------------------
__global__ void ttt_mid_kernel(float* a1, float* a2, float* g1, float* g2, float* dh,
                               const float* __restrict__ fv,
                               const float* __restrict__ lr, int chunk)
{
    const long long n = (long long)NBH * CHUNKN * FD;
    long long idx = (long long)blockIdx.x * 256 + threadIdx.x;
    if (idx >= n) return;
    int bh = (int)(idx / ((long long)CHUNKN * FD));
    int r  = (int)(idx % ((long long)CHUNKN * FD));
    int c = r / FD, e = r % FD;
    int b = bh / NFW, h = bh % NFW;
    int s = chunk * CHUNKN + c;
    long long lbase = ((long long)b * NFW + h) * SEQ + s;
    float l0 = lr[lbase];
    float l1 = lr[(long long)BATCH * NFW * SEQ + lbase];
    float l2 = lr[2LL * BATCH * NFW * SEQ + lbase];

    float A1 = a1[idx], A2 = a2[idx], G1 = g1[idx], G2 = g2[idx], D = dh[idx];
    float hq = siluf(A1) * A2;
    float sig = 1.f / (1.f + __expf(-G1));
    float gate = G1 * sig;
    float hidden = gate * G2;
    float dh1 = D * G2 * (sig * (1.f + G1 * (1.f - sig)));
    float dh2 = D * gate;
    float lv = l1 * fv[((long long)bh * SEQ + s) * FD + e];
    a1[idx] = hq;
    g1[idx] = hidden;
    g2[idx] = l0 * dh1;
    dh[idx] = l2 * dh2;
    a2[idx] = lv;
}

// ---------------------------------------------------------------------------
// Flash attention, causal (window==seq), WMMA for S=QK^T and O=PV.
// Block: 64 q-rows of one (b,head). 256 threads = 8 waves.
// ---------------------------------------------------------------------------
__global__ __launch_bounds__(256) void flash_attn_kernel(
    const float* __restrict__ qa, const float* __restrict__ ka,
    const float* __restrict__ va, float* __restrict__ preO)
{
    const int bh = blockIdx.y;                    // 0..31
    const int qt = blockIdx.x;                    // 0..15
    const int b  = bh / NHEADS;
    const int hN = bh % NHEADS;
    const int tid = threadIdx.x, lane = tid & 31, wave = tid >> 5;
    const int wm = wave & 3;                      // 4 x 16 rows
    const int wn = wave >> 2;                     // S: 2 x 32 cols ; O: 2 x 64 cols
    const int rlo = lane & 15, hi = lane >> 4;

    __shared__ __bf16 Qs[64][136];
    __shared__ __align__(16) char kvraw[128 * 72 * 2];   // max(K 64x136, V 128x72)
    __bf16 (*Ks)[136] = (__bf16 (*)[136])kvraw;
    __bf16 (*Vs)[72]  = (__bf16 (*)[72])kvraw;
    __shared__ float Ss[64][68];
    __shared__ __bf16 Ps[64][72];
    __shared__ float m_i[64], l_i[64], rsc[64];

    const float scale = 0.08838834764831845f;     // 1/sqrt(128)
    const long long qbase = ((long long)bh * SEQ + qt * 64) * HD;

    for (int x = tid * 4; x < 64 * 128; x += 1024) {
        int r = x >> 7, d = x & 127;
        float4 v = *(const float4*)(qa + qbase + (long long)r * HD + d);
        union { uint2 u; __bf16 h[4]; } pk;
        pk.h[0] = to_bf16(v.x); pk.h[1] = to_bf16(v.y);
        pk.h[2] = to_bf16(v.z); pk.h[3] = to_bf16(v.w);
        *(uint2*)&Qs[r][d] = pk.u;
    }
    if (tid < 64) { m_i[tid] = -3.0e38f; l_i[tid] = 0.f; }
    v8f oacc[4];
#pragma unroll
    for (int j = 0; j < 4; ++j)
#pragma unroll
        for (int e = 0; e < 8; ++e) oacc[j][e] = 0.f;
    __syncthreads();

    for (int kt = 0; kt <= qt; ++kt) {
        const long long kbase = ((long long)bh * SEQ + kt * 64) * HD;
        for (int x = tid * 4; x < 64 * 128; x += 1024) {
            int r = x >> 7, d = x & 127;
            float4 v = *(const float4*)(ka + kbase + (long long)r * HD + d);
            union { uint2 u; __bf16 h[4]; } pk;
            pk.h[0] = to_bf16(v.x); pk.h[1] = to_bf16(v.y);
            pk.h[2] = to_bf16(v.z); pk.h[3] = to_bf16(v.w);
            *(uint2*)&Ks[r][d] = pk.u;
        }
        __syncthreads();

        v8f sacc[2];
#pragma unroll
        for (int j = 0; j < 2; ++j)
#pragma unroll
            for (int e = 0; e < 8; ++e) sacc[j][e] = 0.f;
#pragma unroll
        for (int ks = 0; ks < 128; ks += 32) {
            v16bf aq = load_frag(&Qs[wm * 16][ks], 136, lane);
#pragma unroll
            for (int j = 0; j < 2; ++j) {
                v16bf bk = load_frag(&Ks[wn * 32 + j * 16][ks], 136, lane);
                sacc[j] = wmma_bf16(aq, bk, sacc[j]);
            }
        }
#pragma unroll
        for (int j = 0; j < 2; ++j)
#pragma unroll
            for (int r = 0; r < 8; ++r)
                Ss[wm * 16 + hi * 8 + r][wn * 32 + j * 16 + rlo] = sacc[j][r] * scale;
        __syncthreads();

        if (tid < 64) {
            int qrow = qt * 64 + tid;
            float m = m_i[tid];
            float rmax = m;
            for (int j = 0; j < 64; ++j) {
                int kc = kt * 64 + j;
                float sv = (kc <= qrow) ? Ss[tid][j] : -1.0e9f;
                Ss[tid][j] = sv;
                rmax = fmaxf(rmax, sv);
            }
            float sc = __expf(m - rmax);
            float rsum = 0.f;
            for (int j = 0; j < 64; ++j) {
                float p = __expf(Ss[tid][j] - rmax);
                Ps[tid][j] = to_bf16(p);
                rsum += p;
            }
            m_i[tid] = rmax;
            l_i[tid] = l_i[tid] * sc + rsum;
            rsc[tid] = sc;
        }
        __syncthreads();

#pragma unroll
        for (int j = 0; j < 4; ++j)
#pragma unroll
            for (int r = 0; r < 8; ++r)
                oacc[j][r] *= rsc[wm * 16 + hi * 8 + r];
        for (int x = tid * 4; x < 64 * 128; x += 1024) {  // V transposed -> [d][k]
            int r = x >> 7, d = x & 127;
            float4 v = *(const float4*)(va + kbase + (long long)r * HD + d);
            Vs[d + 0][r] = to_bf16(v.x);
            Vs[d + 1][r] = to_bf16(v.y);
            Vs[d + 2][r] = to_bf16(v.z);
            Vs[d + 3][r] = to_bf16(v.w);
        }
        __syncthreads();

#pragma unroll
        for (int ks = 0; ks < 64; ks += 32) {
            v16bf ap = load_frag(&Ps[wm * 16][ks], 72, lane);
#pragma unroll
            for (int j = 0; j < 4; ++j) {
                v16bf bv = load_frag(&Vs[wn * 64 + j * 16][ks], 72, lane);
                oacc[j] = wmma_bf16(ap, bv, oacc[j]);
            }
        }
        __syncthreads();   // before K overwrites kvraw next iter
    }

#pragma unroll
    for (int j = 0; j < 4; ++j) {
        int d = wn * 64 + j * 16 + rlo;
#pragma unroll
        for (int r = 0; r < 8; ++r) {
            int row = wm * 16 + hi * 8 + r;
            int s = qt * 64 + row;
            float ov = oacc[j][r] / l_i[row];
            preO[((long long)b * SEQ + s) * H_DIM + hN * HD + d] = ov;
        }
    }
}

// ---------------------------------------------------------------------------
// o_ttt rmsnorm + add into preO.  One block per (b,h,s).
// ---------------------------------------------------------------------------
__global__ __launch_bounds__(256) void ttt_norm_add_kernel(
    const float* __restrict__ ottt, const float* __restrict__ tnw,
    float* __restrict__ preO)
{
    const int blk = blockIdx.x;          // bh*SEQ + s
    const int bh = blk / SEQ, s = blk % SEQ;
    const int b = bh / NFW, h = bh % NFW;
    const int tid = threadIdx.x;
    const float* rowp = ottt + ((long long)bh * SEQ + s) * FD;
    __shared__ float red[256];
    float v0 = rowp[tid], v1 = rowp[tid + 256];
    red[tid] = v0 * v0 + v1 * v1;
    __syncthreads();
    for (int st = 128; st > 0; st >>= 1) { if (tid < st) red[tid] += red[tid + st]; __syncthreads(); }
    float r = rsqrtf(red[0] * (1.f / FD) + 1e-6f);
    float* outp = preO + ((long long)(b * SEQ + s)) * H_DIM + h * FD;
    outp[tid]       += v0 * r * tnw[tid];
    outp[tid + 256] += v1 * r * tnw[tid + 256];
}

// ---------------------------------------------------------------------------
// launcher
// ---------------------------------------------------------------------------
extern "C" void kernel_launch(void* const* d_in, const int* in_sizes, int n_in,
                              void* d_out, int out_size, void* d_ws, size_t ws_size,
                              hipStream_t stream)
{
    (void)in_sizes; (void)n_in; (void)out_size; (void)ws_size;
    const float* x   = (const float*)d_in[0];
    const float* qw  = (const float*)d_in[1];
    const float* qnw = (const float*)d_in[2];
    const float* knw = (const float*)d_in[3];
    const float* qks = (const float*)d_in[4];
    const float* qko = (const float*)d_in[5];
    const float* lrw = (const float*)d_in[6];
    const float* lrb = (const float*)d_in[7];
    const float* w0  = (const float*)d_in[8];
    const float* w1  = (const float*)d_in[9];
    const float* w2  = (const float*)d_in[10];
    const float* tnw = (const float*)d_in[11];
    const float* opw = (const float*)d_in[12];
    float* out = (float*)d_out;
    float* ws  = (float*)d_ws;

    const long long PH = (long long)FD * FD;          // 262144 per-head matrix
    float* qkv   = ws;                                // 4096*6144
    float* fq    = qkv   + 25165824LL;                // [NBH][SEQ][FD]
    float* fk    = fq    + 8388608LL;
    float* fv    = fk    + 8388608LL;
    float* qa    = fv    + 8388608LL;                 // [B*NH][SEQ][HD]
    float* ka    = qa    + 8388608LL;
    float* va    = ka    + 8388608LL;
    float* lrpre = va    + 8388608LL;                 // 4096*12
    float* lrv   = lrpre + 49152LL;                   // [3][B][NFW][SEQ]
    float* W0    = lrv   + 49152LL;                   // [NBH][FD][FD]
    float* W1    = W0    + NBH * PH;
    float* W2    = W1    + NBH * PH;
    float* a1b   = W2    + NBH * PH;                  // hq after mid
    float* a2b   = a1b   + NBH * PH;                  // lv after mid
    float* g1b   = a2b   + NBH * PH;                  // hidden after mid
    float* g2b   = g1b   + NBH * PH;                  // l0*dh1 after mid
    float* dhb   = g2b   + NBH * PH;                  // l2*dh2 after mid
    float* ottt  = dhb   + NBH * PH;                  // [NBH][SEQ][FD]
    float* preO  = ottt  + 8388608LL;                 // [NTOK][H]

    dim3 blk(256);

    // qkv = x @ qkv_w^T  (4096 x 6144 x 2048)
    wmma_gemm_t<false><<<dim3(48, 32, 1), blk, 0, stream>>>(x, 2048, 1, 0, qw, 2048, 1, 0,
                                                            qkv, 6144, 0, NTOK, 6144, 2048, 0);
    // lrpre = x @ lr_w^T  (4096 x 12 x 2048)  -- needs guards
    wmma_gemm_t<true><<<dim3(1, 32, 1), blk, 0, stream>>>(x, 2048, 1, 0, lrw, 2048, 1, 0,
                                                          lrpre, 12, 0, NTOK, 12, 2048, 0);
    post_qkv_kernel<<<dim3(NTOK), blk, 0, stream>>>(qkv, qnw, knw, qks, qko,
                                                    fq, fk, fv, qa, ka, va);
    lr_post_kernel<<<dim3(192), blk, 0, stream>>>(lrpre, lrb, lrv);
    bcast_w_kernel<<<dim3(1024), blk, 0, stream>>>(w0, W0);
    bcast_w_kernel<<<dim3(1024), blk, 0, stream>>>(w1, W1);
    bcast_w_kernel<<<dim3(1024), blk, 0, stream>>>(w2, W2);

    const long long SB = (long long)SEQ * FD;         // per-head seq stride
    for (int c = 0; c < 4; ++c) {
        const float* fqc = fq + (long long)c * CHUNKN * FD;
        const float* fkc = fk + (long long)c * CHUNKN * FD;
        const float* fvc = fv + (long long)c * CHUNKN * FD;
        dim3 g(4, 4, NBH);
        // a1 = q @ w0^T ; a2 = q @ w2^T ; g1 = k @ w0^T ; g2 = k @ w2^T ; dh = v @ w1
        wmma_gemm_t<false><<<g, blk, 0, stream>>>(fqc, 512, 1, SB, W0, 512, 1, PH, a1b, 512, PH, 512, 512, 512, 0);
        wmma_gemm_t<false><<<g, blk, 0, stream>>>(fqc, 512, 1, SB, W2, 512, 1, PH, a2b, 512, PH, 512, 512, 512, 0);
        wmma_gemm_t<false><<<g, blk, 0, stream>>>(fkc, 512, 1, SB, W0, 512, 1, PH, g1b, 512, PH, 512, 512, 512, 0);
        wmma_gemm_t<false><<<g, blk, 0, stream>>>(fkc, 512, 1, SB, W2, 512, 1, PH, g2b, 512, PH, 512, 512, 512, 0);
        wmma_gemm_t<false><<<g, blk, 0, stream>>>(fvc, 512, 1, SB, W1, 1, 512, PH, dhb, 512, PH, 512, 512, 512, 0);
        ttt_mid_kernel<<<dim3(8192), blk, 0, stream>>>(a1b, a2b, g1b, g2b, dhb, fv, lrv, c);
        // oc = hq @ w1^T  (uses pre-update w1) -> ottt chunk slice
        wmma_gemm_t<false><<<g, blk, 0, stream>>>(a1b, 512, 1, PH, W1, 512, 1, PH,
                                                  ottt + (long long)c * CHUNKN * FD, 512, SB, 512, 512, 512, 0);
        // w1 += lv^T @ hidden ; w0 += (l0*dh1)^T @ k ; w2 += (l2*dh2)^T @ k
        wmma_gemm_t<false><<<g, blk, 0, stream>>>(a2b, 1, 512, PH, g1b, 1, 512, PH, W1, 512, PH, 512, 512, 512, 1);
        wmma_gemm_t<false><<<g, blk, 0, stream>>>(g2b, 1, 512, PH, fkc, 1, 512, SB, W0, 512, PH, 512, 512, 512, 1);
        wmma_gemm_t<false><<<g, blk, 0, stream>>>(dhb, 1, 512, PH, fkc, 1, 512, SB, W2, 512, PH, 512, 512, 512, 1);
    }

    flash_attn_kernel<<<dim3(16, 32), blk, 0, stream>>>(qa, ka, va, preO);
    ttt_norm_add_kernel<<<dim3(NBH * SEQ), blk, 0, stream>>>(ottt, tnw, preO);
    // out = (attn + o_ttt) @ o_proj^T  (4096 x 2048 x 2048)
    wmma_gemm_t<false><<<dim3(16, 32, 1), blk, 0, stream>>>(preO, 2048, 1, 0, opw, 2048, 1, 0,
                                                            out, 2048, 0, NTOK, 2048, 2048, 0);
}
